// scGAT_skipcat_nodropout_PReLU_40106404610259
// MI455X (gfx1250) — matrix operand order repaired and checked by reading, not compiled, not run on hardware
//
#include <hip/hip_runtime.h>

typedef unsigned short u16;
typedef __attribute__((ext_vector_type(16))) __bf16 v16bf;
typedef __attribute__((ext_vector_type(8)))  __bf16 v8bf;
typedef __attribute__((ext_vector_type(8)))  float  v8f;

#define NEG_SLOPE 0.2f

__device__ __forceinline__ u16 f2bf(float f) {
    unsigned u = __float_as_uint(f);
    unsigned r = u + 0x7fffu + ((u >> 16) & 1u);   // round-to-nearest-even
    return (u16)(r >> 16);
}
__device__ __forceinline__ unsigned f2o(float f) {  // monotonic uint encoding
    unsigned u = __float_as_uint(f);
    return (u & 0x80000000u) ? ~u : (u | 0x80000000u);
}
__device__ __forceinline__ float o2f(unsigned u) {
    unsigned v = (u & 0x80000000u) ? (u & 0x7fffffffu) : ~u;
    return __uint_as_float(v);
}
__device__ __forceinline__ float lrelu(float x) { return x > 0.f ? x : NEG_SLOPE * x; }

// ---------------- conversion / transpose ----------------

// x [N,256] f32 -> bf16 into cols [0,256) of xin_b [N,320]
__global__ void k_x_to_bf16(const float* __restrict__ x, u16* __restrict__ dst, int N) {
    int idx = blockIdx.x * blockDim.x + threadIdx.x;
    if (idx >= N * 256) return;
    int i = idx >> 8, c = idx & 255;
    dst[(size_t)i * 320 + c] = f2bf(x[idx]);
}

// W [K,F] f32 row-major -> Wt [F,K] bf16 row-major
__global__ void k_w_transpose(const float* __restrict__ W, u16* __restrict__ Wt, int K, int F) {
    int idx = blockIdx.x * blockDim.x + threadIdx.x;
    if (idx >= K * F) return;
    int k = idx / F, n = idx % F;
    Wt[(size_t)n * K + k] = f2bf(W[idx]);
}

// ---------------- WMMA GEMM: C[M,HC] = A[M,K](bf16,lda) @ Wt[HC,K]^T ----------------
// Each wave computes one 16-row M tile x NT consecutive 16-col N tiles:
// one A fragment load feeds NT WMMAs (A traffic cut NT x, higher wmma/load density).
template <int NT>
__global__ void k_gemm_wmma(const u16* __restrict__ A, int lda,
                            const u16* __restrict__ Bt, int K,
                            float* __restrict__ C, int ldc,
                            int Mtiles, int Ngroups) {
    int wid = blockIdx.x * (blockDim.x >> 5) + (threadIdx.x >> 5);
    if (wid >= Mtiles * Ngroups) return;           // wave-uniform: EXEC stays all-ones
    int tm = wid / Ngroups, tg = wid % Ngroups;
    int lane = threadIdx.x & 31;
    int half = lane >> 4, ln = lane & 15;

    // A 16-bit fragment: elems 0..7 = K half*8+0..7, elems 8..15 = K 16+half*8+0..7
    const u16* arow = A + (size_t)(tm * 16 + ln) * lda + half * 8;
    // B (32x16): elems 0..7 = K half*8+0..7, elems 8..15 = K 16+half*8+0..7; col = ln
    const u16* brow[NT];
#pragma unroll
    for (int t = 0; t < NT; t++)
        brow[t] = Bt + (size_t)((tg * NT + t) * 16 + ln) * K + half * 8;

    v8f acc[NT] = {};
    for (int k0 = 0; k0 < K; k0 += 32) {
        v8bf alo = *(const v8bf*)(arow + k0);
        v8bf ahi = *(const v8bf*)(arow + k0 + 16);
        v16bf a = __builtin_shufflevector(alo, ahi, 0,1,2,3,4,5,6,7,8,9,10,11,12,13,14,15);
#pragma unroll
        for (int t = 0; t < NT; t++) {
            v8bf blo = *(const v8bf*)(brow[t] + k0);
            v8bf bhi = *(const v8bf*)(brow[t] + k0 + 16);
            v16bf b = __builtin_shufflevector(blo, bhi, 0,1,2,3,4,5,6,7,8,9,10,11,12,13,14,15);
            acc[t] = __builtin_amdgcn_wmma_f32_16x16x32_bf16(false, a, false, b, (short)0,
                                                             acc[t], false, false);
        }
    }
    // C fragment: VGPR r -> row half*8+r, col ln
    float* cbase = C + (size_t)(tm * 16 + half * 8) * ldc + ln;
#pragma unroll
    for (int t = 0; t < NT; t++) {
        float* cp = cbase + (tg * NT + t) * 16;
#pragma unroll
        for (int r = 0; r < 8; r++) cp[(size_t)r * ldc] = acc[t][r];
    }
}

// ---------------- attention scores + self-loop max init ----------------
__global__ void k_scores_init(const float* __restrict__ hbuf,
                              const float* __restrict__ atts, const float* __restrict__ attd,
                              float* __restrict__ as_, float* __restrict__ ad_,
                              unsigned* __restrict__ mu, int N, int Cdim) {
    int idx = blockIdx.x * blockDim.x + threadIdx.x;
    if (idx >= N * 8) return;
    int i = idx >> 3, h = idx & 7;
    const float* hp = hbuf + (size_t)i * 8 * Cdim + h * Cdim;
    const float* sa = atts + h * Cdim;
    const float* da = attd + h * Cdim;
    float ss = 0.f, sd = 0.f;
    for (int c = 0; c < Cdim; c++) { ss += hp[c] * sa[c]; sd += hp[c] * da[c]; }
    as_[idx] = ss;
    ad_[idx] = sd;
    mu[idx] = f2o(lrelu(ss + sd));                 // self-loop seeds the segment max
}

// ---------------- edge pass 1: segment max ----------------
__global__ void k_edge_max(const int* __restrict__ src, const int* __restrict__ dst,
                           const float* __restrict__ as_, const float* __restrict__ ad_,
                           unsigned* __restrict__ mu, int E) {
    int e = blockIdx.x * blockDim.x + threadIdx.x;
    if (e >= E) return;
    int s = src[e], d = dst[e];
    const float4* s4 = (const float4*)(as_ + (size_t)s * 8);
    const float4* d4 = (const float4*)(ad_ + (size_t)d * 8);
    float4 a0 = s4[0], a1 = s4[1], b0 = d4[0], b1 = d4[1];
    float v[8] = {a0.x + b0.x, a0.y + b0.y, a0.z + b0.z, a0.w + b0.w,
                  a1.x + b1.x, a1.y + b1.y, a1.z + b1.z, a1.w + b1.w};
    unsigned* mp = mu + (size_t)d * 8;
#pragma unroll
    for (int h = 0; h < 8; h++) atomicMax(mp + h, f2o(lrelu(v[h])));
}

// ---------------- decode max, seed sum with self-loop ----------------
__global__ void k_finalize_max(const float* __restrict__ as_, const float* __restrict__ ad_,
                               const unsigned* __restrict__ mu, float* __restrict__ mf,
                               float* __restrict__ sb, int NH) {
    int idx = blockIdx.x * blockDim.x + threadIdx.x;
    if (idx >= NH) return;
    float m = o2f(mu[idx]);
    mf[idx] = m;
    sb[idx] = expf(lrelu(as_[idx] + ad_[idx]) - m);
}

// ---------------- edge pass 2: segment sum ----------------
__global__ void k_edge_sum(const int* __restrict__ src, const int* __restrict__ dst,
                           const float* __restrict__ as_, const float* __restrict__ ad_,
                           const float* __restrict__ mf, float* __restrict__ sb, int E) {
    int e = blockIdx.x * blockDim.x + threadIdx.x;
    if (e >= E) return;
    int s = src[e], d = dst[e];
    const float4* s4 = (const float4*)(as_ + (size_t)s * 8);
    const float4* d4 = (const float4*)(ad_ + (size_t)d * 8);
    const float4* m4 = (const float4*)(mf + (size_t)d * 8);
    float4 a0 = s4[0], a1 = s4[1], b0 = d4[0], b1 = d4[1], m0 = m4[0], m1 = m4[1];
    float v[8] = {a0.x + b0.x, a0.y + b0.y, a0.z + b0.z, a0.w + b0.w,
                  a1.x + b1.x, a1.y + b1.y, a1.z + b1.z, a1.w + b1.w};
    float mm[8] = {m0.x, m0.y, m0.z, m0.w, m1.x, m1.y, m1.z, m1.w};
    float* sp = sb + (size_t)d * 8;
#pragma unroll
    for (int h = 0; h < 8; h++) atomicAdd(sp + h, expf(lrelu(v[h]) - mm[h]));
}

// ---------------- init output with self-loop message (full overwrite) ----------------
__global__ void k_init_out(const float* __restrict__ hbuf,
                           const float* __restrict__ as_, const float* __restrict__ ad_,
                           const float* __restrict__ mf, const float* __restrict__ sb,
                           float* __restrict__ obuf, int N, int Cdim) {
    int idx = blockIdx.x * blockDim.x + threadIdx.x;
    if (idx >= N * 8) return;
    int i = idx >> 3, h = idx & 7;
    float alpha = expf(lrelu(as_[idx] + ad_[idx]) - mf[idx]) / (sb[idx] + 1e-16f);
    const float* hp = hbuf + (size_t)i * 8 * Cdim + h * Cdim;
    float* op = obuf + (size_t)i * 8 * Cdim + h * Cdim;
    for (int c = 0; c < Cdim; c++) op[c] = hp[c] * alpha;
}

// ---------------- edge pass 3: weighted message aggregation ----------------
__global__ void k_edge_aggr(const int* __restrict__ src, const int* __restrict__ dst,
                            const float* __restrict__ as_, const float* __restrict__ ad_,
                            const float* __restrict__ mf, const float* __restrict__ sb,
                            const float* __restrict__ hbuf, float* __restrict__ obuf,
                            int E, int Cdim) {
    int idx = blockIdx.x * blockDim.x + threadIdx.x;
    if (idx >= E * 8) return;
    int e = idx % E;            // consecutive threads -> consecutive edges (coalesced)
    int h = idx / E;
    int s = src[e], d = dst[e];
    float v = lrelu(as_[(size_t)s * 8 + h] + ad_[(size_t)d * 8 + h]);
    float alpha = expf(v - mf[(size_t)d * 8 + h]) / (sb[(size_t)d * 8 + h] + 1e-16f);
    const float* hp = hbuf + (size_t)s * 8 * Cdim + h * Cdim;
    float* op = obuf + (size_t)d * 8 * Cdim + h * Cdim;
    for (int c = 0; c < Cdim; c++) atomicAdd(op + c, hp[c] * alpha);
}

// ---------------- bias + PReLU -> bf16 (layers 1,2) ----------------
__global__ void k_epi12(const float* __restrict__ obuf, const float* __restrict__ bias,
                        const float* __restrict__ p, u16* __restrict__ dstb,
                        int dstride, int col0, int N) {
    int idx = blockIdx.x * blockDim.x + threadIdx.x;
    if (idx >= N * 64) return;
    int i = idx >> 6, j = idx & 63;
    float v = obuf[idx] + bias[j];
    float a = p[0];
    v = v >= 0.f ? v : a * v;
    dstb[(size_t)i * dstride + col0 + j] = f2bf(v);
}

// ---------------- head-mean + bias + log_softmax (layer 3) ----------------
__global__ void k_epi3(const float* __restrict__ obuf, const float* __restrict__ b3,
                       float* __restrict__ out, int N) {
    int i = blockIdx.x * blockDim.x + threadIdx.x;
    if (i >= N) return;
    const float* op = obuf + (size_t)i * 128;
    float t[16];
#pragma unroll
    for (int c = 0; c < 16; c++) {
        float s = 0.f;
#pragma unroll
        for (int h = 0; h < 8; h++) s += op[h * 16 + c];
        t[c] = s * 0.125f + b3[c];
    }
    float mx = t[0];
#pragma unroll
    for (int c = 1; c < 16; c++) mx = fmaxf(mx, t[c]);
    float sum = 0.f;
#pragma unroll
    for (int c = 0; c < 16; c++) sum += expf(t[c] - mx);
    float ls = logf(sum);
#pragma unroll
    for (int c = 0; c < 16; c++) out[(size_t)i * 16 + c] = t[c] - mx - ls;
}

// =====================================================================

extern "C" void kernel_launch(void* const* d_in, const int* in_sizes, int n_in,
                              void* d_out, int out_size, void* d_ws, size_t ws_size,
                              hipStream_t stream) {
    const float* x   = (const float*)d_in[0];
    const int*   ei  = (const int*)d_in[1];
    const float* W1  = (const float*)d_in[2];
    const float* as1 = (const float*)d_in[3];
    const float* ad1 = (const float*)d_in[4];
    const float* b1  = (const float*)d_in[5];
    const float* W2  = (const float*)d_in[6];
    const float* as2 = (const float*)d_in[7];
    const float* ad2 = (const float*)d_in[8];
    const float* b2  = (const float*)d_in[9];
    const float* W3  = (const float*)d_in[10];
    const float* as3 = (const float*)d_in[11];
    const float* ad3 = (const float*)d_in[12];
    const float* b3  = (const float*)d_in[13];
    const float* p1  = (const float*)d_in[14];
    const float* p2  = (const float*)d_in[15];
    float* out = (float*)d_out;

    const int N = in_sizes[0] / 256;
    const int E = in_sizes[1] / 2;
    const int* srcp = ei;
    const int* dstp = ei + E;

    // ---- carve workspace (256B aligned slabs) ----
    char* w = (char*)d_ws;
    auto alloc = [&](size_t b) -> char* { char* p = w; w += (b + 255) & ~(size_t)255; return p; };
    u16*      xinb = (u16*)alloc((size_t)N * 320 * sizeof(u16));   // [x | prelu(h1)] bf16
    u16*      w1t  = (u16*)alloc((size_t)64 * 256 * sizeof(u16));
    u16*      w2t  = (u16*)alloc((size_t)64 * 320 * sizeof(u16));
    u16*      w3t  = (u16*)alloc((size_t)128 * 64 * sizeof(u16));
    u16*      h2b  = (u16*)alloc((size_t)N * 64 * sizeof(u16));    // prelu(h2) bf16
    float*    hbuf = (float*)alloc((size_t)N * 128 * sizeof(float));
    float*    obuf = (float*)alloc((size_t)N * 128 * sizeof(float));
    float*    asb  = (float*)alloc((size_t)N * 8 * sizeof(float));
    float*    adb  = (float*)alloc((size_t)N * 8 * sizeof(float));
    unsigned* mub  = (unsigned*)alloc((size_t)N * 8 * sizeof(unsigned));
    float*    mfb  = (float*)alloc((size_t)N * 8 * sizeof(float));
    float*    sbb  = (float*)alloc((size_t)N * 8 * sizeof(float));

    const int TB = 256;
    auto nb = [](long long n, int tb) { return (unsigned)((n + tb - 1) / tb); };

    // ---- one GAT layer (WMMA GEMM + segment-softmax + aggregation) ----
    auto run_layer = [&](const u16* Ab, int lda, int K, const u16* Bt, int Cdim,
                         const float* atts, const float* attd) {
        const int HC = 8 * Cdim;
        const int Mtiles = N / 16;
        const int Ngroups = (HC / 16) / 4;         // 4 N-tiles per wave
        const int waves = Mtiles * Ngroups;
        k_gemm_wmma<4><<<nb(waves, 4), 128, 0, stream>>>(Ab, lda, Bt, K, hbuf, HC,
                                                         Mtiles, Ngroups);
        k_scores_init<<<nb((long long)N * 8, TB), TB, 0, stream>>>(hbuf, atts, attd, asb, adb,
                                                                   mub, N, Cdim);
        k_edge_max<<<nb(E, TB), TB, 0, stream>>>(srcp, dstp, asb, adb, mub, E);
        k_finalize_max<<<nb((long long)N * 8, TB), TB, 0, stream>>>(asb, adb, mub, mfb, sbb,
                                                                    N * 8);
        k_edge_sum<<<nb(E, TB), TB, 0, stream>>>(srcp, dstp, asb, adb, mfb, sbb, E);
        k_init_out<<<nb((long long)N * 8, TB), TB, 0, stream>>>(hbuf, asb, adb, mfb, sbb, obuf,
                                                                N, Cdim);
        k_edge_aggr<<<nb((long long)E * 8, TB), TB, 0, stream>>>(srcp, dstp, asb, adb, mfb, sbb,
                                                                 hbuf, obuf, E, Cdim);
    };

    // ---- prep: bf16 conversions + weight transposes ----
    k_x_to_bf16<<<nb((long long)N * 256, TB), TB, 0, stream>>>(x, xinb, N);
    k_w_transpose<<<nb(256 * 64, TB), TB, 0, stream>>>(W1, w1t, 256, 64);
    k_w_transpose<<<nb(320 * 64, TB), TB, 0, stream>>>(W2, w2t, 320, 64);
    k_w_transpose<<<nb(64 * 128, TB), TB, 0, stream>>>(W3, w3t, 64, 128);

    // ---- layer 1: x[N,256] -> h1[N,64]; prelu -> xinb cols 256..319 ----
    run_layer(xinb, 320, 256, w1t, 8, as1, ad1);
    k_epi12<<<nb((long long)N * 64, TB), TB, 0, stream>>>(obuf, b1, p1, xinb, 320, 256, N);

    // ---- layer 2: [x|h1][N,320] -> h2[N,64]; prelu -> h2b ----
    run_layer(xinb, 320, 320, w2t, 8, as2, ad2);
    k_epi12<<<nb((long long)N * 64, TB), TB, 0, stream>>>(obuf, b2, p2, h2b, 64, 0, N);

    // ---- layer 3: h2[N,64] -> [N,8,16]; head-mean + bias + log_softmax ----
    run_layer(h2b, 64, 64, w3t, 16, as3, ad3);
    k_epi3<<<nb(N, TB), TB, 0, stream>>>(obuf, b3, out, N);
}
